// TransformerBlock_8340826489292
// MI455X (gfx1250) — matrix operand order repaired and checked by reading, not compiled, not run on hardware
//
#include <hip/hip_runtime.h>
#include <math.h>

#define TT   2048
#define DD   1024
#define HH   16
#define DHH  64
#define DFFN 4096
#define BB   2

#define BM 128
#define BN 128
#define BK 64            // staged K-depth: 2 WMMA k-steps per barrier
#define LDSW (BK + 8)    // ushorts per LDS row: 64 data + 8 pad (bank spread)

typedef __attribute__((ext_vector_type(16))) __bf16 v16bf;
typedef __attribute__((ext_vector_type(8)))  float  v8f;
typedef __attribute__((ext_vector_type(4)))  unsigned int u32x4;

union Frag16 {
    v16bf bf;
    u32x4 q[2];
};

__device__ __forceinline__ unsigned short f2bf(float f) {
    unsigned int u = __builtin_bit_cast(unsigned int, f);
    u += 0x7fffu + ((u >> 16) & 1u);           // round-to-nearest-even
    return (unsigned short)(u >> 16);
}

__device__ __forceinline__ v8f wmma_bf16(const Frag16& a, const Frag16& b, v8f c) {
    return __builtin_amdgcn_wmma_f32_16x16x32_bf16(
        false, a.bf, false, b.bf, (short)0, c, false, false);
}

// CDNA5 async global->LDS 16B copy, tracked by ASYNCcnt (ISA 08_async_tensor §4)
__device__ __forceinline__ void async_copy16(const unsigned short* g, unsigned int lds_off) {
    asm volatile("global_load_async_to_lds_b128 %0, %1, off"
                 :: "v"(lds_off), "v"((unsigned long long)(size_t)g)
                 : "memory");
}
__device__ __forceinline__ void wait_async0() {
    asm volatile("s_wait_asynccnt 0x0" ::: "memory");
}

// A-matrix 16x32 bf16 fragment from global: lane m=lane%16, 16 contiguous K, base=16*(lane>=16)
__device__ __forceinline__ void load_fragA(Frag16& f, const unsigned short* base,
                                           int ld, int m0, int k0, int lane) {
    const int row = m0 + (lane & 15);
    const int kb  = k0 + ((lane >> 4) << 4);
    const unsigned short* p = base + (size_t)row * ld + kb;
    f.q[0] = *reinterpret_cast<const u32x4*>(p);
    f.q[1] = *reinterpret_cast<const u32x4*>(p + 8);
}

// B-matrix 32x16 bf16 fragment from global (B[k][n] = W[n][k]): 16 contiguous K
__device__ __forceinline__ void load_fragB(Frag16& f, const unsigned short* base,
                                           int ld, int n0, int k0, int lane) {
    const int row = n0 + (lane & 15);
    const int kb  = k0 + ((lane >> 4) << 4);
    const unsigned short* p = base + (size_t)row * ld + kb;
    f.q[0] = *reinterpret_cast<const u32x4*>(p);
    f.q[1] = *reinterpret_cast<const u32x4*>(p + 8);
}

// ---------------------------------------------------------------- utilities
__global__ void cvt_bf16_kernel(const float* __restrict__ in,
                                unsigned short* __restrict__ out, int n) {
    int i = blockIdx.x * blockDim.x + threadIdx.x;
    int stride = gridDim.x * blockDim.x;
    for (; i < n; i += stride) out[i] = f2bf(in[i]);
}

__global__ __launch_bounds__(256) void ln_kernel(const float* __restrict__ x,
                                                 const float* __restrict__ g,
                                                 const float* __restrict__ b,
                                                 unsigned short* __restrict__ out) {
    __shared__ float rs[256], rss[256];
    const int row = blockIdx.x;
    const int tid = threadIdx.x;
    const float* xr = x + (size_t)row * DD;
    float v[4], s = 0.f, ss = 0.f;
#pragma unroll
    for (int i = 0; i < 4; ++i) {
        v[i] = xr[tid + i * 256];
        s  += v[i];
        ss += v[i] * v[i];
    }
    rs[tid] = s; rss[tid] = ss;
    __syncthreads();
    for (int off = 128; off > 0; off >>= 1) {
        if (tid < off) { rs[tid] += rs[tid + off]; rss[tid] += rss[tid + off]; }
        __syncthreads();
    }
    const float mean = rs[0] * (1.f / DD);
    const float var  = rss[0] * (1.f / DD) - mean * mean;
    const float rinv = rsqrtf(var + 1e-5f);
#pragma unroll
    for (int i = 0; i < 4; ++i) {
        const int col = tid + i * 256;
        out[(size_t)row * DD + col] = f2bf((v[i] - mean) * rinv * g[col] + b[col]);
    }
}

// --------------------------------------------- LDS-staged double-buffered GEMM
// C[M,N] = A[M,K] x W[N,K]^T + bias, 128x128 block tile, async global->LDS,
// BK=64 staged (16 WMMAs per workgroup barrier).
// MODE 0: outF = acc+bias+resid (f32)
// MODE 1: outB = bf16(gelu_exact(acc+bias))
// MODE 2: QKV scatter -> qout (x0.125, [B,H,T,DH]), kout ([B,H,T,DH]), vtout ([B,H,DH,T])
template <int MODE>
__global__ __launch_bounds__(256) void gemm_lds_kernel(
    const unsigned short* __restrict__ A, const unsigned short* __restrict__ W,
    const float* __restrict__ bias, int K, int N,
    const float* __restrict__ resid, float* __restrict__ outF,
    unsigned short* __restrict__ outB,
    unsigned short* __restrict__ qout, unsigned short* __restrict__ kout,
    unsigned short* __restrict__ vtout)
{
    __shared__ __align__(16) unsigned short Abuf[2][BM * LDSW];
    __shared__ __align__(16) unsigned short Bbuf[2][BN * LDSW];

    const int tid   = threadIdx.x;
    const int lane  = tid & 31;
    const int wave  = tid >> 5;
    const int m_blk = blockIdx.y * BM;
    const int n_blk = blockIdx.x * BN;
    const int wm    = (wave >> 1) * 32;   // 0/32/64/96
    const int wn    = (wave & 1) * 64;    // 0/64

    // stage one 128x64 A tile + 128x64 B tile into LDS buffer `buf`
    // 1024 chunks of 16B per tile, 4 per thread per tile
    auto stage = [&](int buf, int k0) {
        const unsigned int abase = (unsigned int)(size_t)&Abuf[buf][0];
        const unsigned int bbase = (unsigned int)(size_t)&Bbuf[buf][0];
#pragma unroll
        for (int i = 0; i < 4; ++i) {
            const int c   = tid + i * 256;
            const int row = c >> 3;             // 8 chunks per 128B row
            const int seg = c & 7;
            const unsigned int loff = (unsigned int)(row * LDSW + seg * 8) * 2u;
            async_copy16(A + (size_t)(m_blk + row) * K + k0 + seg * 8, abase + loff);
            async_copy16(W + (size_t)(n_blk + row) * K + k0 + seg * 8, bbase + loff);
        }
    };

    v8f acc[2][4] = {};

    stage(0, 0);
    wait_async0();
    __syncthreads();

    int cur = 0;
    for (int k0 = 0; k0 < K; k0 += BK) {
        if (k0 + BK < K) stage(cur ^ 1, k0 + BK);   // prefetch next tile (ASYNCcnt)

        const int r   = lane & 15;
        const int kb  = (lane >> 4) << 4;   // A: +0 / +16
        const int kb2 = (lane >> 4) << 4;   // B: +0 / +16
#pragma unroll
        for (int kk = 0; kk < BK; kk += 32) {
            Frag16 a[2], b[4];
#pragma unroll
            for (int mi = 0; mi < 2; ++mi) {
                const unsigned short* p = &Abuf[cur][(wm + mi * 16 + r) * LDSW + kk + kb];
                a[mi].q[0] = *reinterpret_cast<const u32x4*>(p);
                a[mi].q[1] = *reinterpret_cast<const u32x4*>(p + 8);
            }
#pragma unroll
            for (int j = 0; j < 4; ++j) {
                const unsigned short* p = &Bbuf[cur][(wn + j * 16 + r) * LDSW + kk + kb2];
                b[j].q[0] = *reinterpret_cast<const u32x4*>(p);
                b[j].q[1] = *reinterpret_cast<const u32x4*>(p + 8);
            }
#pragma unroll
            for (int mi = 0; mi < 2; ++mi)
#pragma unroll
                for (int j = 0; j < 4; ++j)
                    acc[mi][j] = wmma_bf16(a[mi], b[j], acc[mi][j]);
        }

        wait_async0();
        __syncthreads();
        cur ^= 1;
    }

    // ------------------------------------------------------------- epilogue
#pragma unroll
    for (int mi = 0; mi < 2; ++mi) {
#pragma unroll
        for (int j = 0; j < 4; ++j) {
            const int col  = n_blk + wn + j * 16 + (lane & 15);
            const float bv = bias[col];
#pragma unroll
            for (int r = 0; r < 8; ++r) {
                const int rowg = m_blk + wm + mi * 16 + r + ((lane >> 4) << 3);
                const float v  = acc[mi][j][r] + bv;
                if constexpr (MODE == 0) {
                    outF[(size_t)rowg * N + col] = v + resid[(size_t)rowg * N + col];
                } else if constexpr (MODE == 1) {
                    const float gelu = 0.5f * v * (1.0f + erff(v * 0.70710678118654752f));
                    outB[(size_t)rowg * N + col] = f2bf(gelu);
                } else {
                    const int which  = col >> 10;        // 0=q 1=k 2=v
                    const int within = col & 1023;
                    const int head   = within >> 6;
                    const int d      = within & 63;
                    const int bI     = rowg >> 11;       // rowg / T
                    const int t      = rowg & (TT - 1);
                    if (which == 0)
                        qout[(((size_t)(bI * HH + head) * TT) + t) * DHH + d] = f2bf(v * 0.125f);
                    else if (which == 1)
                        kout[(((size_t)(bI * HH + head) * TT) + t) * DHH + d] = f2bf(v);
                    else
                        vtout[(((size_t)(bI * HH + head) * DHH) + d) * TT + t] = f2bf(v);
                }
            }
        }
    }
}

// ------------------------------------------------------- flash attention
// 1 wave = 16 query rows; online softmax; 32-key chunks; causal.
__global__ __launch_bounds__(128) void attn_kernel(
    const unsigned short* __restrict__ qb, const unsigned short* __restrict__ kb,
    const unsigned short* __restrict__ vt, unsigned short* __restrict__ attn)
{
    __shared__ __align__(16) unsigned short psh[4][16 * 32];
    const int lane = threadIdx.x & 31;
    const int wave = threadIdx.x >> 5;
    const int bh   = blockIdx.y;
    const int q0   = (blockIdx.x * 4 + wave) * 16;
    const int bI   = bh / HH, h = bh % HH;

    const unsigned short* Q  = qb + (size_t)bh * TT * DHH;
    const unsigned short* Km = kb + (size_t)bh * TT * DHH;
    const unsigned short* Vt = vt + (size_t)bh * DHH * TT;

    Frag16 qa0, qa1;
    load_fragA(qa0, Q, DHH, q0, 0, lane);
    load_fragA(qa1, Q, DHH, q0, 32, lane);

    v8f o[4] = {};
    float mrow[8], lrow[8];
#pragma unroll
    for (int r = 0; r < 8; ++r) { mrow[r] = -3.0e38f; lrow[r] = 0.f; }

    const int qlast = q0 + 15;
    unsigned short* pw = psh[wave];

    for (int kt0 = 0; kt0 <= qlast; kt0 += 32) {
        v8f s[2] = {};
#pragma unroll
        for (int nh = 0; nh < 2; ++nh) {
            Frag16 kf0, kf1;
            load_fragB(kf0, Km, DHH, kt0 + nh * 16, 0, lane);
            load_fragB(kf1, Km, DHH, kt0 + nh * 16, 32, lane);
            s[nh] = wmma_bf16(qa0, kf0, s[nh]);
            s[nh] = wmma_bf16(qa1, kf1, s[nh]);
        }
        const int c0 = kt0 + (lane & 15);
        const int c1 = c0 + 16;
#pragma unroll
        for (int r = 0; r < 8; ++r) {
            const int mr   = r + ((lane >> 4) << 3);
            const int qrow = q0 + mr;
            float s0 = (c0 <= qrow) ? s[0][r] : -3.0e38f;
            float s1 = (c1 <= qrow) ? s[1][r] : -3.0e38f;
            float mx = fmaxf(s0, s1);
#pragma unroll
            for (int sh = 1; sh < 16; sh <<= 1)
                mx = fmaxf(mx, __shfl_xor(mx, sh, 32));
            const float mnew  = fmaxf(mrow[r], mx);
            const float alpha = __expf(mrow[r] - mnew);
            const float p0 = __expf(s0 - mnew);
            const float p1 = __expf(s1 - mnew);
            float ps = p0 + p1;
#pragma unroll
            for (int sh = 1; sh < 16; sh <<= 1)
                ps += __shfl_xor(ps, sh, 32);
            lrow[r] = lrow[r] * alpha + ps;
            mrow[r] = mnew;
#pragma unroll
            for (int j = 0; j < 4; ++j) o[j][r] *= alpha;
            pw[mr * 32 + (lane & 15)]      = f2bf(p0);
            pw[mr * 32 + 16 + (lane & 15)] = f2bf(p1);
        }
        // C-layout -> A-layout via per-wave LDS tile (row-major 16x32 bf16)
        Frag16 pf;
        {
            const int rowm = lane & 15;
            const int kbv  = (lane >> 4) << 4;
            const unsigned short* pp = pw + rowm * 32 + kbv;
            pf.q[0] = *reinterpret_cast<const u32x4*>(pp);
            pf.q[1] = *reinterpret_cast<const u32x4*>(pp + 8);
        }
#pragma unroll
        for (int j = 0; j < 4; ++j) {
            Frag16 vf;
            const int drow = j * 16 + (lane & 15);
            const unsigned short* pv = Vt + (size_t)drow * TT + kt0 + ((lane >> 4) << 4);
            vf.q[0] = *reinterpret_cast<const u32x4*>(pv);
            vf.q[1] = *reinterpret_cast<const u32x4*>(pv + 8);
            o[j] = wmma_bf16(pf, vf, o[j]);
        }
    }

#pragma unroll
    for (int j = 0; j < 4; ++j) {
#pragma unroll
        for (int r = 0; r < 8; ++r) {
            const int mr    = r + ((lane >> 4) << 3);
            const int token = q0 + mr;
            const float v = o[j][r] / lrow[r];
            attn[(((size_t)(bI * TT + token)) * DD) + h * DHH + j * 16 + (lane & 15)] = f2bf(v);
        }
    }
}

// ---------------------------------------------------------------- launcher
extern "C" void kernel_launch(void* const* d_in, const int* in_sizes, int n_in,
                              void* d_out, int out_size, void* d_ws, size_t ws_size,
                              hipStream_t stream) {
    (void)in_sizes; (void)n_in; (void)out_size; (void)ws_size;
    const float* x     = (const float*)d_in[0];
    const float* ln1_g = (const float*)d_in[1];
    const float* ln1_b = (const float*)d_in[2];
    const float* ln2_g = (const float*)d_in[3];
    const float* ln2_b = (const float*)d_in[4];
    const float* w_qkv = (const float*)d_in[5];
    const float* b_qkv = (const float*)d_in[6];
    const float* w_out = (const float*)d_in[7];
    const float* b_out = (const float*)d_in[8];
    const float* w1    = (const float*)d_in[9];
    const float* b1    = (const float*)d_in[10];
    const float* w2    = (const float*)d_in[11];
    const float* b2    = (const float*)d_in[12];
    float* out = (float*)d_out;

    char* ws = (char*)d_ws;
    size_t off = 0;
    auto alloc = [&](size_t bytes) {
        char* p = ws + off;
        off += (bytes + 255) & ~(size_t)255;
        return p;
    };
    const int M = BB * TT;  // 4096 tokens
    unsigned short* w_qkv_bf = (unsigned short*)alloc((size_t)3 * DD * DD * 2);
    unsigned short* w_out_bf = (unsigned short*)alloc((size_t)DD * DD * 2);
    unsigned short* w1_bf    = (unsigned short*)alloc((size_t)DFFN * DD * 2);
    unsigned short* w2_bf    = (unsigned short*)alloc((size_t)DD * DFFN * 2);
    unsigned short* h_bf     = (unsigned short*)alloc((size_t)M * DD * 2);   // reused for h2
    unsigned short* q_bf     = (unsigned short*)alloc((size_t)BB * HH * TT * DHH * 2);
    unsigned short* k_bf     = (unsigned short*)alloc((size_t)BB * HH * TT * DHH * 2);
    unsigned short* vt_bf    = (unsigned short*)alloc((size_t)BB * HH * DHH * TT * 2);
    unsigned short* attn_bf  = (unsigned short*)alloc((size_t)M * DD * 2);
    float*          x2       = (float*)alloc((size_t)M * DD * 4);
    unsigned short* ff_bf    = (unsigned short*)alloc((size_t)M * DFFN * 2);

    cvt_bf16_kernel<<<512, 256, 0, stream>>>(w_qkv, w_qkv_bf, 3 * DD * DD);
    cvt_bf16_kernel<<<512, 256, 0, stream>>>(w_out, w_out_bf, DD * DD);
    cvt_bf16_kernel<<<512, 256, 0, stream>>>(w1, w1_bf, DFFN * DD);
    cvt_bf16_kernel<<<512, 256, 0, stream>>>(w2, w2_bf, DD * DFFN);

    ln_kernel<<<M, 256, 0, stream>>>(x, ln1_g, ln1_b, h_bf);

    gemm_lds_kernel<2><<<dim3(3 * DD / BN, M / BM), 256, 0, stream>>>(
        h_bf, w_qkv_bf, b_qkv, DD, 3 * DD, nullptr, nullptr, nullptr,
        q_bf, k_bf, vt_bf);

    attn_kernel<<<dim3(TT / 64, BB * HH), 128, 0, stream>>>(q_bf, k_bf, vt_bf, attn_bf);

    gemm_lds_kernel<0><<<dim3(DD / BN, M / BM), 256, 0, stream>>>(
        attn_bf, w_out_bf, b_out, DD, DD, x, x2, nullptr, nullptr, nullptr, nullptr);

    ln_kernel<<<M, 256, 0, stream>>>(x2, ln2_g, ln2_b, h_bf);

    gemm_lds_kernel<1><<<dim3(DFFN / BN, M / BM), 256, 0, stream>>>(
        h_bf, w1_bf, b1, DD, DFFN, nullptr, nullptr, ff_bf, nullptr, nullptr, nullptr);

    gemm_lds_kernel<0><<<dim3(DD / BN, M / BM), 256, 0, stream>>>(
        ff_bf, w2_bf, b2, DFFN, DD, x2, out, nullptr, nullptr, nullptr, nullptr);
}
